// Seq2seq_20409684591145
// MI455X (gfx1250) — matrix operand order repaired and checked by reading, not compiled, not run on hardware
//
#include <hip/hip_runtime.h>
#include <stdint.h>

// Problem constants (from the reference)
#define SOS_TOK 1
#define EOS_TOK 2
#define MAXLEN  256
#define VV      32000
#define EE      512
#define UU      1024
#define Z4      4096          // 4*U
#define SS      256           // encoder sequence length
#define KIN     (EE + UU)     // 1536 = K of the LSTM GEMV ([x;h] vs [W;R])
#define NKE     (KIN / 32)    // 48 K-chunks for LSTM gemv
#define NKD     (UU / 32)     // 32 K-chunks for logits gemv
#define ZTILES  (Z4 / 16)     // 256 N-tiles of 16 (gates)
#define VTILES  (VV / 16)     // 2000 N-tiles of 16 (vocab)

typedef __attribute__((ext_vector_type(16))) __bf16          bf16x16;
typedef __attribute__((ext_vector_type(8)))  unsigned short  ushort8;
typedef __attribute__((ext_vector_type(8)))  float           floatx8;

struct U16x16 { ushort8 lo, hi; };   // 32B, bit_cast'able to bf16x16

struct __align__(256) Hdr {
  unsigned           bar_cnt;
  unsigned           bar_gen;
  unsigned long long argmax;   // (monotone-float-key << 32) | (0xFFFFFFFF - index)
  int                tok;      // current decoder input token
  int                done;
};

__device__ __forceinline__ unsigned short f2bf(float f) {
  unsigned u = __float_as_uint(f);
  unsigned r = 0x7FFFu + ((u >> 16) & 1u);          // round-to-nearest-even
  return (unsigned short)((u + r) >> 16);
}
__device__ __forceinline__ float sigf(float x) { return 1.0f / (1.0f + __expf(-x)); }

// ---------------------------------------------------------------------------
// Grid-wide sense-reversing barrier (re-initialized by init kernel per launch).
// ---------------------------------------------------------------------------
__device__ __forceinline__ void grid_sync(Hdr* hdr) {
  __threadfence();
  __syncthreads();
  if (threadIdx.x == 0) {
    unsigned nb = gridDim.x;
    unsigned g  = __hip_atomic_load(&hdr->bar_gen, __ATOMIC_ACQUIRE, __HIP_MEMORY_SCOPE_AGENT);
    unsigned a  = __hip_atomic_fetch_add(&hdr->bar_cnt, 1u, __ATOMIC_ACQ_REL, __HIP_MEMORY_SCOPE_AGENT);
    if (a == nb - 1u) {
      __hip_atomic_store(&hdr->bar_cnt, 0u, __ATOMIC_RELAXED, __HIP_MEMORY_SCOPE_AGENT);
      __hip_atomic_fetch_add(&hdr->bar_gen, 1u, __ATOMIC_RELEASE, __HIP_MEMORY_SCOPE_AGENT);
    } else {
      while (__hip_atomic_load(&hdr->bar_gen, __ATOMIC_ACQUIRE, __HIP_MEMORY_SCOPE_AGENT) == g)
        __builtin_amdgcn_s_sleep(1);
    }
  }
  __syncthreads();
}

// ---------------------------------------------------------------------------
// Pipelined M=1 dot-product core over `nc` K-chunks of 32.
//  p  : pre-swizzled B fragments, 1KB per chunk, lane-offset already applied
//  ab : per-lane A base (lane0 -> s_vec, lane16 -> s_vec+16, others -> zeros)
// ---------------------------------------------------------------------------
__device__ __forceinline__ floatx8 dot_chunks(const unsigned short* __restrict__ p,
                                              const unsigned short* __restrict__ ab,
                                              int nc) {
  floatx8 acc = {};
  bf16x16 b   = *(const bf16x16*)(p);
  ushort8 alo = *(const ushort8*)(ab);
  ushort8 ahi = *(const ushort8*)(ab + 8);
  for (int kc = 0; kc < nc - 1; ++kc) {
    // issue next-chunk loads before this chunk's WMMA (latency overlap)
    bf16x16 bn   = *(const bf16x16*)(p + (size_t)(kc + 1) * 512);
    ushort8 alon = *(const ushort8*)(ab + (kc + 1) * 32);
    ushort8 ahin = *(const ushort8*)(ab + (kc + 1) * 32 + 8);
    __builtin_prefetch((const void*)(p + (size_t)(kc + 5) * 512), 0, 1);
    U16x16 cat; cat.lo = alo; cat.hi = ahi;
    bf16x16 a = __builtin_bit_cast(bf16x16, cat);
    acc = __builtin_amdgcn_wmma_f32_16x16x32_bf16(false, a, false, b,
                                                  (short)0, acc, false, false);
    b = bn; alo = alon; ahi = ahin;
  }
  U16x16 cat; cat.lo = alo; cat.hi = ahi;
  bf16x16 a = __builtin_bit_cast(bf16x16, cat);
  acc = __builtin_amdgcn_wmma_f32_16x16x32_bf16(false, a, false, b,
                                                (short)0, acc, false, false);
  return acc;
}

// ---------------------------------------------------------------------------
// Gates GEMV, K split in half across wave pairs so all 512 waves are busy
// (2*ZTILES = 512 work units). Partial sums go to out0/out1; the activation
// phase adds out0 + out1 + bias.
// ---------------------------------------------------------------------------
__device__ __forceinline__ void gemv_wmma_ksplit(const unsigned short* __restrict__ Wfrag,
                                                 const unsigned short* __restrict__ abase,
                                                 int nK, int ntiles,
                                                 float* __restrict__ out0,
                                                 float* __restrict__ out1,
                                                 int wave, int nwaves, int lane) {
  const int half = nK >> 1;
  for (int unit = wave; unit < ntiles * 2; unit += nwaves) {
    const int t  = unit >> 1;
    const int hs = unit & 1;
    const unsigned short* p  = Wfrag + ((size_t)t * nK + (size_t)hs * half) * 512 + lane * 16;
    const unsigned short* ab = abase + hs * half * 32;
    floatx8 acc = dot_chunks(p, ab, half);
    float* out = hs ? out1 : out0;
    if (lane < 16) out[t * 16 + lane] = acc[0];     // row M=0
  }
}

// ---------------------------------------------------------------------------
// logits = h · dense_W + dense_b, fused argmax: per-wave shuffle reduction of
// a monotone (float-key, inverted-index) pack, one atomicMax per tile.
// ---------------------------------------------------------------------------
__device__ __forceinline__ void logits_argmax(const unsigned short* __restrict__ Wfrag,
                                              const unsigned short* __restrict__ abase,
                                              const float* __restrict__ dense_b,
                                              unsigned long long* __restrict__ slot,
                                              int wave, int nwaves, int lane) {
  for (int t = wave; t < VTILES; t += nwaves) {
    const unsigned short* p = Wfrag + (size_t)t * NKD * 512 + lane * 16;
    floatx8 acc = dot_chunks(p, abase, NKD);
    int   n = t * 16 + (lane & 15);
    float v = (lane < 16) ? (acc[0] + dense_b[n]) : -__builtin_inff();
    unsigned ub  = __float_as_uint(v);
    unsigned key = (ub & 0x80000000u) ? ~ub : (ub | 0x80000000u);  // monotone map
    unsigned lo  = 0xFFFFFFFFu - (unsigned)n;                      // ties -> lowest idx
    unsigned hi  = key;
#pragma unroll
    for (int off = 8; off >= 1; off >>= 1) {
      unsigned olo = __shfl_xor(lo, off, 16);
      unsigned ohi = __shfl_xor(hi, off, 16);
      unsigned long long o   = ((unsigned long long)ohi << 32) | olo;
      unsigned long long cur = ((unsigned long long)hi  << 32) | lo;
      if (o > cur) { lo = olo; hi = ohi; }
    }
    if (lane == 0) atomicMax(slot, ((unsigned long long)hi << 32) | lo);
  }
}

// ---------------------------------------------------------------------------
// Weight pack: f32 [K,N] (optionally split W(0..Ksplit)/R(Ksplit..)) -> bf16
// WMMA-B fragment layout dst[((t*nK + kc)*32 + lane)*16 + e],
// with k = kc*32 + (lane>>4)*16 + e, n = t*16 + (lane&15).
// ---------------------------------------------------------------------------
__global__ void __launch_bounds__(256) pack_weights(const float* __restrict__ W,
                                                    const float* __restrict__ R,
                                                    int Ksplit, int nK, int ntiles, int N,
                                                    unsigned short* __restrict__ dst) {
  size_t total  = (size_t)ntiles * nK * 512;
  size_t stride = (size_t)gridDim.x * blockDim.x;
  for (size_t idx = (size_t)blockIdx.x * blockDim.x + threadIdx.x; idx < total; idx += stride) {
    int    e   = (int)(idx & 15);
    int    l   = (int)((idx >> 4) & 31);
    size_t rem = idx >> 9;
    int    kc  = (int)(rem % (size_t)nK);
    int    t   = (int)(rem / (size_t)nK);
    int    k   = kc * 32 + ((l >> 4) << 4) + e;
    int    n   = t * 16 + (l & 15);
    float  v   = (k < Ksplit) ? W[(size_t)k * N + n] : R[(size_t)(k - Ksplit) * N + n];
    dst[idx] = f2bf(v);
  }
}

// ---------------------------------------------------------------------------
// Per-launch state init: barrier, argmax slot, token state, h=c=0, first vec.
// ---------------------------------------------------------------------------
__global__ void __launch_bounds__(256) seq2seq_init(const int* __restrict__ x,
                                                    const float* __restrict__ emb_enc,
                                                    unsigned short* __restrict__ vec,
                                                    unsigned short* __restrict__ hb,
                                                    float* __restrict__ h,
                                                    float* __restrict__ c, Hdr* hdr) {
  int g = blockIdx.x * blockDim.x + threadIdx.x;
  if (g == 0) {
    hdr->bar_cnt = 0; hdr->bar_gen = 0; hdr->argmax = 0ull;
    hdr->tok = SOS_TOK; hdr->done = 0;
  }
  if (g < UU) { h[g] = 0.f; c[g] = 0.f; vec[EE + g] = 0; hb[g] = 0; }
  if (g < EE) vec[g] = f2bf(emb_enc[(size_t)x[0] * EE + g]);
}

// ---------------------------------------------------------------------------
// Persistent cooperative kernel: 256 encoder + 256 decoder steps.
// ---------------------------------------------------------------------------
__global__ void __launch_bounds__(256, 1) seq2seq_main(
    const int* __restrict__ x,
    const float* __restrict__ emb_enc, const float* __restrict__ emb_dec,
    const float* __restrict__ enc_b,   const float* __restrict__ dec_b,
    const float* __restrict__ dense_b,
    const unsigned short* __restrict__ encW,
    const unsigned short* __restrict__ decW,
    const unsigned short* __restrict__ denseW,
    unsigned short* __restrict__ vec,  // bf16 [emb(512); h(1024)]
    unsigned short* __restrict__ hb,   // bf16 h2 for logits A
    float* __restrict__ h, float* __restrict__ c,
    float* __restrict__ zp0, float* __restrict__ zp1,   // K-split partial gates
    Hdr* hdr, int* __restrict__ out_tok) {
  const int lane   = threadIdx.x & 31;
  const int gtid   = blockIdx.x * blockDim.x + threadIdx.x;
  const int nthr   = gridDim.x * blockDim.x;
  const int wave   = gtid >> 5;
  const int nwaves = nthr >> 5;
  __shared__ unsigned short s_vec[KIN];
  __shared__ unsigned short s_zero[KIN + 32];   // all-zero A source for lanes != 0,16
  __shared__ int s_tok;

  // per-lane fixed A-fragment base (see dot_chunks)
  const unsigned short* abase =
      (lane == 0) ? s_vec : (lane == 16) ? (s_vec + 16) : s_zero;

  for (int i = threadIdx.x; i < KIN + 32; i += blockDim.x) s_zero[i] = 0;

  // ---------------- encoder ----------------
  for (int t = 0; t < SS; ++t) {
    for (int i = threadIdx.x; i < KIN; i += blockDim.x) s_vec[i] = vec[i];
    __syncthreads();
    gemv_wmma_ksplit(encW, abase, NKE, ZTILES, zp0, zp1, wave, nwaves, lane);
    grid_sync(hdr);

    const bool m = (x[t] != 0);
    for (int u = gtid; u < UU; u += nthr) {
      float zi = zp0[u]          + zp1[u]          + enc_b[u];
      float zf = zp0[UU + u]     + zp1[UU + u]     + enc_b[UU + u];
      float zg = zp0[2 * UU + u] + zp1[2 * UU + u] + enc_b[2 * UU + u];
      float zo = zp0[3 * UU + u] + zp1[3 * UU + u] + enc_b[3 * UU + u];
      float cc = c[u];
      float c2 = sigf(zf) * cc + sigf(zi) * tanhf(zg);
      float h2 = sigf(zo) * tanhf(c2);
      float hn = m ? h2 : h[u];
      float cn = m ? c2 : cc;
      h[u] = hn; c[u] = cn;
      vec[EE + u] = f2bf(hn);
    }
    for (int e = gtid; e < EE; e += nthr) {
      float ev = (t + 1 < SS) ? emb_enc[(size_t)x[t + 1] * EE + e]
                              : emb_dec[(size_t)SOS_TOK * EE + e];
      vec[e] = f2bf(ev);
    }
    grid_sync(hdr);
  }

  // ---------------- decoder ----------------
  for (int t = 0; t < MAXLEN; ++t) {
    for (int i = threadIdx.x; i < KIN; i += blockDim.x) s_vec[i] = vec[i];
    __syncthreads();
    gemv_wmma_ksplit(decW, abase, NKE, ZTILES, zp0, zp1, wave, nwaves, lane);
    grid_sync(hdr);

    const int  tok = ((volatile Hdr*)hdr)->tok;
    const bool m   = (tok != 0);
    for (int u = gtid; u < UU; u += nthr) {
      float zi = zp0[u]          + zp1[u]          + dec_b[u];
      float zf = zp0[UU + u]     + zp1[UU + u]     + dec_b[UU + u];
      float zg = zp0[2 * UU + u] + zp1[2 * UU + u] + dec_b[2 * UU + u];
      float zo = zp0[3 * UU + u] + zp1[3 * UU + u] + dec_b[3 * UU + u];
      float cc = c[u];
      float c2 = sigf(zf) * cc + sigf(zi) * tanhf(zg);
      float h2 = sigf(zo) * tanhf(c2);
      float hn = m ? h2 : h[u];
      float cn = m ? c2 : cc;
      h[u] = hn; c[u] = cn;
      unsigned short hbf = f2bf(hn);
      hb[u] = hbf; vec[EE + u] = hbf;
    }
    grid_sync(hdr);

    for (int i = threadIdx.x; i < UU; i += blockDim.x) s_vec[i] = hb[i];
    __syncthreads();
    logits_argmax(denseW, abase, dense_b, &hdr->argmax, wave, nwaves, lane);
    grid_sync(hdr);

    if (blockIdx.x == 0) {      // finalize: new token, output, next embedding
      if (threadIdx.x == 0) {
        unsigned long long pk = ((volatile Hdr*)hdr)->argmax;
        int n  = (int)(0xFFFFFFFFu - (unsigned)(pk & 0xFFFFFFFFull));
        int dn = ((volatile Hdr*)hdr)->done;
        out_tok[t] = dn ? 0 : n;                       // out uses *old* done
        ((volatile Hdr*)hdr)->done   = dn | (n == EOS_TOK);
        ((volatile Hdr*)hdr)->tok    = n;
        ((volatile Hdr*)hdr)->argmax = 0ull;
        s_tok = n;
      }
      __syncthreads();
      const int nt = s_tok;
      for (int e = threadIdx.x; e < EE; e += blockDim.x)
        vec[e] = f2bf(emb_dec[(size_t)nt * EE + e]);
    }
    grid_sync(hdr);
  }
}

// ---------------------------------------------------------------------------
extern "C" void kernel_launch(void* const* d_in, const int* in_sizes, int n_in,
                              void* d_out, int out_size, void* d_ws, size_t ws_size,
                              hipStream_t stream) {
  (void)in_sizes; (void)n_in; (void)out_size; (void)ws_size;
  const int*   x       = (const int*)  d_in[0];
  const float* emb_enc = (const float*)d_in[1];
  const float* enc_W   = (const float*)d_in[2];
  const float* enc_R   = (const float*)d_in[3];
  const float* enc_b   = (const float*)d_in[4];
  const float* emb_dec = (const float*)d_in[5];
  const float* dec_W   = (const float*)d_in[6];
  const float* dec_R   = (const float*)d_in[7];
  const float* dec_b   = (const float*)d_in[8];
  const float* dense_W = (const float*)d_in[9];
  const float* dense_b = (const float*)d_in[10];

  char*  ws  = (char*)d_ws;
  size_t off = 0;
  auto alloc = [&](size_t bytes) -> char* {
    char* p = ws + off;
    off = (off + bytes + 255) & ~(size_t)255;
    return p;
  };
  const size_t PAD = 16384;  // prefetch overrun slack past each weight block
  Hdr*            hdr    = (Hdr*)           alloc(sizeof(Hdr));
  unsigned short* vec    = (unsigned short*)alloc((size_t)KIN * 2);
  unsigned short* hb     = (unsigned short*)alloc((size_t)UU * 2);
  float*          h      = (float*)         alloc((size_t)UU * 4);
  float*          c      = (float*)         alloc((size_t)UU * 4);
  float*          zp0    = (float*)         alloc((size_t)Z4 * 4);
  float*          zp1    = (float*)         alloc((size_t)Z4 * 4);
  unsigned short* encW   = (unsigned short*)alloc((size_t)ZTILES * NKE * 512 * 2 + PAD); // 12.6 MB
  unsigned short* decW   = (unsigned short*)alloc((size_t)ZTILES * NKE * 512 * 2 + PAD); // 12.6 MB
  unsigned short* denseW = (unsigned short*)alloc((size_t)VTILES * NKD * 512 * 2 + PAD); // 65.5 MB

  dim3 blk(256);
  // One-time (per launch) weight pack: f32 -> bf16, WMMA fragment layout.
  pack_weights<<<2048, blk, 0, stream>>>(enc_W,   enc_R,   EE, NKE, ZTILES, Z4, encW);
  pack_weights<<<2048, blk, 0, stream>>>(dec_W,   dec_R,   EE, NKE, ZTILES, Z4, decW);
  pack_weights<<<8192, blk, 0, stream>>>(dense_W, dense_W, UU, NKD, VTILES, VV, denseW);
  seq2seq_init<<<4, blk, 0, stream>>>(x, emb_enc, vec, hb, h, c, hdr);

  const int NB = 64;  // persistent cooperative grid (64 WGs x 8 wave32)
  seq2seq_main<<<NB, blk, 0, stream>>>(x, emb_enc, emb_dec, enc_b, dec_b, dense_b,
                                       encW, decW, denseW, vec, hb, h, c, zp0, zp1,
                                       hdr, (int*)d_out);
}
